// BitLinear_1949915152491
// MI455X (gfx1250) — matrix operand order repaired and checked by reading, not compiled, not run on hardware
//
#include <hip/hip_runtime.h>
#include <hip/hip_bf16.h>

typedef __bf16 bf16;
typedef __attribute__((ext_vector_type(16))) __bf16 v16bf;
typedef __attribute__((ext_vector_type(8)))  __bf16 v8bf;
typedef __attribute__((ext_vector_type(4)))  __bf16 v4bf;
typedef __attribute__((ext_vector_type(8)))  float  v8f;
typedef __attribute__((ext_vector_type(4)))  float  v4f;
typedef __attribute__((ext_vector_type(4)))  int    v4i;

#define EPSV 1e-5f
#define GEMM_M 8192
#define GEMM_N 4096
#define GEMM_K 4096

// ---------------------------------------------------------------------------
// CDNA5 async global->LDS copy (ASYNCcnt-tracked). Guarded so the file always
// compiles; fallback stages through generic stores (same pipeline structure).
// Builtin prototype (probe-confirmed by round-2 diagnostics):
//   (v4i addrspace(1)* src, v4i addrspace(3)* dst, const int off, const int cpol)
// s_wait_asynccnt requires a literal immediate -> macro call sites.
// ---------------------------------------------------------------------------
#if defined(__gfx1250__) &&                                          \
    __has_builtin(__builtin_amdgcn_global_load_async_to_lds_b128) && \
    __has_builtin(__builtin_amdgcn_s_wait_asynccnt)
#define USE_ASYNC 1
#define ASYNC_WAIT(n) __builtin_amdgcn_s_wait_asynccnt(n)
#else
#define USE_ASYNC 0
#define ASYNC_WAIT(n) ((void)0)
#endif

__device__ __forceinline__ void cp16_g2l(const bf16* g, bf16* l) {
#if USE_ASYNC
  __builtin_amdgcn_global_load_async_to_lds_b128(
      (__attribute__((address_space(1))) v4i*)g,
      (__attribute__((address_space(3))) v4i*)l, 0, 0);
#else
  *(v8bf*)l = *(const v8bf*)g;
#endif
}

// ---------------------------------------------------------------------------
// Pass 1a: partial sums of |W| (deterministic tree reduction, no atomics)
// ---------------------------------------------------------------------------
__global__ void abssum_partial(const float* __restrict__ W,
                               float* __restrict__ partial, int n4) {
  __shared__ float red[256];
  float s = 0.f;
  int stride = gridDim.x * blockDim.x;
  for (int i = blockIdx.x * blockDim.x + threadIdx.x; i < n4; i += stride) {
    v4f w = ((const v4f*)W)[i];
    s += fabsf(w.x) + fabsf(w.y) + fabsf(w.z) + fabsf(w.w);
  }
  red[threadIdx.x] = s;
  __syncthreads();
  for (int off = 128; off > 0; off >>= 1) {
    if (threadIdx.x < off) red[threadIdx.x] += red[threadIdx.x + off];
    __syncthreads();
  }
  if (threadIdx.x == 0) partial[blockIdx.x] = red[0];
}

// ---------------------------------------------------------------------------
// Pass 1b: reduce partials -> scale = mean(|W|)
// ---------------------------------------------------------------------------
__global__ void abssum_final(const float* __restrict__ partial,
                             float* __restrict__ scale, int nparts,
                             float inv_count) {
  __shared__ float red[256];
  float s = 0.f;
  for (int i = threadIdx.x; i < nparts; i += 256) s += partial[i];
  red[threadIdx.x] = s;
  __syncthreads();
  for (int off = 128; off > 0; off >>= 1) {
    if (threadIdx.x < off) red[threadIdx.x] += red[threadIdx.x + off];
    __syncthreads();
  }
  if (threadIdx.x == 0) scale[0] = red[0] * inv_count;
}

// ---------------------------------------------------------------------------
// Pass 2: Wq = bf16(round(W / (scale+eps)))  — small integers, exact in bf16
// ---------------------------------------------------------------------------
__global__ void quantize_w(const float* __restrict__ W,
                           const float* __restrict__ scale_p,
                           bf16* __restrict__ Wq) {
  float inv = 1.f / (scale_p[0] + EPSV);
  int i = (blockIdx.x * 256 + threadIdx.x) * 4;
  v4f w = *(const v4f*)(W + i);
  v4bf q = { (bf16)rintf(w.x * inv), (bf16)rintf(w.y * inv),
             (bf16)rintf(w.z * inv), (bf16)rintf(w.w * inv) };
  *(v4bf*)(Wq + i) = q;
}

// ---------------------------------------------------------------------------
// Pass 3: x -> bf16 (makes full GEMM working set L2-resident: ~100MB < 192MB)
// ---------------------------------------------------------------------------
__global__ void cvt_x(const float* __restrict__ X, bf16* __restrict__ Xb) {
  int i = (blockIdx.x * 256 + threadIdx.x) * 4;
  v4f w = *(const v4f*)(X + i);
  v4bf q = { (bf16)w.x, (bf16)w.y, (bf16)w.z, (bf16)w.w };
  *(v4bf*)(Xb + i) = q;
}

// ---------------------------------------------------------------------------
// Pass 4: WMMA GEMM  Y[m][n] = scale * sum_k A[m][k]*B[n][k]
//   Block tile 128x128, BK=64; 8 waves (wave32), wave tile 32x64
//   = 2x4 v_wmma_f32_16x16x32_bf16 per 32-wide K-step.
//   LDS row stride 72 bf16 (144B = 36 banks) -> conflict-free ds_load_b128.
//   Double-buffered LDS + async global->LDS copies: one-tile software
//   pipeline (issue k+1, s_wait_asynccnt 8, barrier, compute k).
// ---------------------------------------------------------------------------
#define BM 128
#define BN 128
#define BK 64
#define LDSTR 72
#define NT (GEMM_K / BK)

__global__ __launch_bounds__(256, 2) void bitlinear_gemm(
    const bf16* __restrict__ A, const bf16* __restrict__ B,
    const float* __restrict__ scale_p, float* __restrict__ Y) {
  __shared__ bf16 lds_a[2][BM * LDSTR];
  __shared__ bf16 lds_b[2][BN * LDSTR];

  const int tid  = threadIdx.x;
  const int lane = tid & 31;
  const int wave = tid >> 5;     // 0..7
  const int waveM = wave & 3;    // 4 wave rows  (32 rows each)
  const int waveN = wave >> 2;   // 2 wave cols  (64 cols each)
  const int l16   = lane & 15;
  const int halfA = (lane < 16) ? 0 : 8;   // 16-bit A frag: upper lanes K+8
  const int halfB = (lane < 16) ? 0 : 8;   // 16-bit B frag: upper lanes K+8

  const int m0 = blockIdx.y * BM;
  const int n0 = blockIdx.x * BN;

  // per-thread copy chunks: 4x 16B for A tile + 4x 16B for B tile
  const int crow = tid >> 3;          // rows tid/8, tid/8+32, +64, +96
  const int ccol = (tid & 7) * 8;     // 8-elem column chunk

  v8f acc[2][4] = {};

  // issue the global->LDS copies of K-tile kt into LDS buffer `buf`
  auto issue_tile = [&](int kt, int buf) {
    const int k0 = kt * BK;
#pragma unroll
    for (int i = 0; i < 4; ++i) {
      int row = crow + i * 32;
      cp16_g2l(A + (size_t)(m0 + row) * GEMM_K + k0 + ccol,
               &lds_a[buf][row * LDSTR + ccol]);
      cp16_g2l(B + (size_t)(n0 + row) * GEMM_K + k0 + ccol,
               &lds_b[buf][row * LDSTR + ccol]);
    }
  };

  issue_tile(0, 0);

  for (int kt = 0; kt < NT; ++kt) {
    const int cur = kt & 1;
    if (kt + 1 < NT) {
      issue_tile(kt + 1, cur ^ 1);  // prefetch next tile into other buffer
      ASYNC_WAIT(8);                // tile kt resident (loads retire in order)
    } else {
      ASYNC_WAIT(0);
    }
    __syncthreads();

    const bf16* la = lds_a[cur];
    const bf16* lb = lds_b[cur];
#pragma unroll
    for (int kk = 0; kk < BK; kk += 32) {
      v16bf afrag[2], bfrag[4];
#pragma unroll
      for (int mi = 0; mi < 2; ++mi) {
        int r = waveM * 32 + mi * 16 + l16;
        // A 16x32 bf16 frag: v0-3 = K[0..7]+half, v4-7 = K[16..23]+half
        v8bf lo = *(const v8bf*)(la + r * LDSTR + kk + halfA);
        v8bf hi = *(const v8bf*)(la + r * LDSTR + kk + 16 + halfA);
        afrag[mi] = __builtin_shufflevector(lo, hi, 0, 1, 2, 3, 4, 5, 6, 7, 8,
                                            9, 10, 11, 12, 13, 14, 15);
      }
#pragma unroll
      for (int ni = 0; ni < 4; ++ni) {
        int rn = waveN * 64 + ni * 16 + l16;
        // B 32x16 frag: lane holds column N=rn, v0-3 = K[0..7]+half, v4-7 = K[16..23]+half
        v8bf lo = *(const v8bf*)(lb + rn * LDSTR + kk + halfB);
        v8bf hi = *(const v8bf*)(lb + rn * LDSTR + kk + 16 + halfB);
        bfrag[ni] = __builtin_shufflevector(lo, hi, 0, 1, 2, 3, 4, 5, 6, 7, 8,
                                            9, 10, 11, 12, 13, 14, 15);
      }
#pragma unroll
      for (int mi = 0; mi < 2; ++mi)
#pragma unroll
        for (int ni = 0; ni < 4; ++ni)
          acc[mi][ni] = __builtin_amdgcn_wmma_f32_16x16x32_bf16(
              false, afrag[mi], false, bfrag[ni], (short)0, acc[mi][ni],
              false, false);
    }
    __syncthreads();  // all waves done reading buf[cur] before it is refilled
  }

  // ---- epilogue: y *= mean(|W|), store fp32 ----
  const float s = scale_p[0];
#pragma unroll
  for (int mi = 0; mi < 2; ++mi) {
    int rbase = m0 + waveM * 32 + mi * 16 + ((lane < 16) ? 0 : 8);
#pragma unroll
    for (int ni = 0; ni < 4; ++ni) {
      int col = n0 + waveN * 64 + ni * 16 + l16;
#pragma unroll
      for (int r = 0; r < 8; ++r)
        Y[(size_t)(rbase + r) * GEMM_N + col] = acc[mi][ni][r] * s;
    }
  }
}

// ---------------------------------------------------------------------------
extern "C" void kernel_launch(void* const* d_in, const int* in_sizes, int n_in,
                              void* d_out, int out_size, void* d_ws,
                              size_t ws_size, hipStream_t stream) {
  const float* x = (const float*)d_in[0];   // (4,2048,4096) fp32
  const float* W = (const float*)d_in[1];   // (4096,4096)   fp32
  float* y = (float*)d_out;                 // (4,2048,4096) fp32

  const int nW = GEMM_N * GEMM_K;           // 16,777,216
  const int nX = GEMM_M * GEMM_K;           // 33,554,432

  // workspace layout (offsets 256-aligned):
  //   [0, 4KB)         : 1024 float partial sums
  //   [4KB, 4KB+4)     : final scale
  //   [8KB, +32MB)     : Wq bf16
  //   [8KB+32MB, +64MB): x bf16
  float* partial = (float*)d_ws;
  float* scale   = (float*)((char*)d_ws + 4096);
  bf16*  Wq      = (bf16*)((char*)d_ws + 8192);
  bf16*  Xb      = (bf16*)((char*)d_ws + 8192 + (size_t)nW * sizeof(bf16));

  abssum_partial<<<1024, 256, 0, stream>>>(W, partial, nW / 4);
  abssum_final<<<1, 256, 0, stream>>>(partial, scale, 1024, 1.f / (float)nW);
  quantize_w<<<nW / (256 * 4), 256, 0, stream>>>(W, scale, Wq);
  cvt_x<<<nX / (256 * 4), 256, 0, stream>>>(x, Xb);

  dim3 grid(GEMM_N / BN, GEMM_M / BM);  // 32 x 64
  bitlinear_gemm<<<grid, 256, 0, stream>>>(Xb, Wq, scale, y);
}
